// TemporalGCN_42657615184081
// MI455X (gfx1250) — matrix operand order repaired and checked by reading, not compiled, not run on hardware
//
#include <hip/hip_runtime.h>

// ---------------- problem constants ----------------
#define BATCH    64
#define C_IN     16
#define T_IN     8192
#define T1       4096          // after pool1
#define C1       16
#define C2       32
#define T2       2048          // after pool2
#define HID      64
#define OUTC     10
#define N_NODES  (BATCH * T2)  // 131072
#define N_EDGES  2097152

typedef __attribute__((ext_vector_type(16))) __bf16 v16bf;
typedef __attribute__((ext_vector_type(8)))  float  v8f;

union ABReg { uint4 q[2]; v16bf v; };

__device__ __forceinline__ unsigned short f2bf(float f) {
    union { float f; unsigned int u; } x; x.f = f;
    unsigned int u = x.u;
    unsigned int r = u + 0x7FFFu + ((u >> 16) & 1u);   // round-to-nearest-even
    return (unsigned short)(r >> 16);
}

// ---------------- conv1 + relu + maxpool2 ----------------
// x [64,16,8192] f32 -> h1 [64,16,4096] f32
__global__ void conv1_kernel(const float* __restrict__ x,
                             const float* __restrict__ w,   // [16,16,5]
                             const float* __restrict__ bias,
                             float* __restrict__ out) {
    int tp = blockIdx.x * blockDim.x + threadIdx.x;   // 0..4095
    int co = blockIdx.y;                              // 0..15
    int b  = blockIdx.z;                              // 0..63
    if (tp >= T1) return;
    int t0 = tp * 2;
    const float* xb = x + (size_t)b * C_IN * T_IN;
    const float* wc = w + co * C_IN * 5;
    float acc0 = bias[co], acc1 = bias[co];
    for (int ci = 0; ci < C_IN; ++ci) {
        const float* xc = xb + (size_t)ci * T_IN;
        float v[6];
        #pragma unroll
        for (int k = 0; k < 6; ++k) {
            int t = t0 - 2 + k;
            v[k] = (t >= 0 && t < T_IN) ? xc[t] : 0.0f;
        }
        const float* wk = wc + ci * 5;
        #pragma unroll
        for (int k = 0; k < 5; ++k) {
            acc0 = fmaf(v[k],     wk[k], acc0);
            acc1 = fmaf(v[k + 1], wk[k], acc1);
        }
    }
    out[((size_t)b * C1 + co) * T1 + tp] = fmaxf(fmaxf(acc0, acc1), 0.0f);
}

// ---------------- conv2 + relu + maxpool2 -> node-major bf16 feat ----------------
// h1 [64,16,4096] -> feat bf16 [N_NODES, 32], node = b*2048 + tp
__global__ void conv2_kernel(const float* __restrict__ h1,
                             const float* __restrict__ w,   // [32,16,5]
                             const float* __restrict__ bias,
                             unsigned short* __restrict__ feat) {
    int tp = blockIdx.x * blockDim.x + threadIdx.x;   // 0..2047
    int co = blockIdx.y;                              // 0..31
    int b  = blockIdx.z;
    if (tp >= T2) return;
    int t0 = tp * 2;
    const float* xb = h1 + (size_t)b * C1 * T1;
    const float* wc = w + co * C1 * 5;
    float acc0 = bias[co], acc1 = bias[co];
    for (int ci = 0; ci < C1; ++ci) {
        const float* xc = xb + (size_t)ci * T1;
        float v[6];
        #pragma unroll
        for (int k = 0; k < 6; ++k) {
            int t = t0 - 2 + k;
            v[k] = (t >= 0 && t < T1) ? xc[t] : 0.0f;
        }
        const float* wk = wc + ci * 5;
        #pragma unroll
        for (int k = 0; k < 5; ++k) {
            acc0 = fmaf(v[k],     wk[k], acc0);
            acc1 = fmaf(v[k + 1], wk[k], acc1);
        }
    }
    float r = fmaxf(fmaxf(acc0, acc1), 0.0f);
    feat[((size_t)b * T2 + tp) * C2 + co] = f2bf(r);
}

// ---------------- pack GCN weights into WMMA B register image ----------------
// B image used here (contiguous k order, same as the A loads): VGPR j, lane L: N = n0+(L&15), K = kbase + (L>>4)*16 + 2j,2j+1
// p1: W1 [32,64] -> 4 col tiles * 32 lanes * 8 dwords
// p2: W2 [64,64] -> (kh*4+ct) tiles * 32 lanes * 8 dwords
__global__ void pack_kernel(const float* __restrict__ W1,
                            const float* __restrict__ W2,
                            unsigned int* __restrict__ p1,
                            unsigned int* __restrict__ p2) {
    int tid = threadIdx.x;   // 384 threads
    if (tid < 128) {
        int ct = tid >> 5, lane = tid & 31;
        int half = lane >> 4, n = ct * 16 + (lane & 15);
        #pragma unroll
        for (int j = 0; j < 8; ++j) {
            int k0 = half * 16 + 2 * j;
            unsigned int lo = f2bf(W1[k0 * HID + n]);
            unsigned int hi = f2bf(W1[(k0 + 1) * HID + n]);
            p1[ct * 256 + lane * 8 + j] = lo | (hi << 16);
        }
    } else {
        int t = tid - 128;            // 0..255
        int tile = t >> 5;            // kh*4 + ct
        int lane = t & 31;
        int kh = tile >> 2, ct = tile & 3;
        int half = lane >> 4, n = ct * 16 + (lane & 15);
        #pragma unroll
        for (int j = 0; j < 8; ++j) {
            int k0 = kh * 32 + half * 16 + 2 * j;
            unsigned int lo = f2bf(W2[k0 * HID + n]);
            unsigned int hi = f2bf(W2[(k0 + 1) * HID + n]);
            p2[tile * 256 + lane * 8 + j] = lo | (hi << 16);
        }
    }
}

// ---------------- WMMA GEMM: xw = feat[N,32] @ W1[32,64] ----------------
__global__ void gemm_n32_kernel(const unsigned short* __restrict__ featb,
                                const unsigned int* __restrict__ packB,
                                float* __restrict__ xw) {
    int tid  = threadIdx.x;
    int lane = tid & 31;
    int wave = blockIdx.x * (blockDim.x >> 5) + (tid >> 5);
    int node0 = wave * 16;
    if (node0 >= N_NODES) return;
    int half = lane >> 4;
    int nl   = lane & 15;

    ABReg bt[4];
    #pragma unroll
    for (int ct = 0; ct < 4; ++ct) {
        const uint4* p = (const uint4*)(packB + ct * 256 + lane * 8);
        bt[ct].q[0] = p[0];
        bt[ct].q[1] = p[1];
    }
    // A: 16x32 bf16; per lane two contiguous b128 loads
    ABReg a;
    const unsigned short* rowp = featb + (size_t)(node0 + nl) * 32 + half * 16;
    a.q[0] = *(const uint4*)(rowp);
    a.q[1] = *(const uint4*)(rowp + 8);

    #pragma unroll
    for (int ct = 0; ct < 4; ++ct) {
        v8f c = {};
        c = __builtin_amdgcn_wmma_f32_16x16x32_bf16(false, a.v, false, bt[ct].v,
                                                    (short)0, c, false, false);
        float* op = xw + (size_t)(node0 + half * 8) * HID + ct * 16 + nl;
        #pragma unroll
        for (int r = 0; r < 8; ++r) op[(size_t)r * HID] = c[r];
    }
}

// ---------------- WMMA GEMM: xw = g1[N,64] @ W2[64,64] (2-step K accumulate) ----------------
__global__ void gemm_n64_kernel(const unsigned short* __restrict__ g1b,
                                const unsigned int* __restrict__ packB,
                                float* __restrict__ xw) {
    int tid  = threadIdx.x;
    int lane = tid & 31;
    int wave = blockIdx.x * (blockDim.x >> 5) + (tid >> 5);
    int node0 = wave * 16;
    if (node0 >= N_NODES) return;
    int half = lane >> 4;
    int nl   = lane & 15;

    ABReg a0, a1;
    const unsigned short* rowp = g1b + (size_t)(node0 + nl) * HID;
    a0.q[0] = *(const uint4*)(rowp + half * 16);
    a0.q[1] = *(const uint4*)(rowp + 8 + half * 16);
    a1.q[0] = *(const uint4*)(rowp + 32 + half * 16);
    a1.q[1] = *(const uint4*)(rowp + 40 + half * 16);

    #pragma unroll
    for (int ct = 0; ct < 4; ++ct) {
        ABReg b0, b1;
        const uint4* p0 = (const uint4*)(packB + (0 * 4 + ct) * 256 + lane * 8);
        const uint4* p1 = (const uint4*)(packB + (1 * 4 + ct) * 256 + lane * 8);
        b0.q[0] = p0[0]; b0.q[1] = p0[1];
        b1.q[0] = p1[0]; b1.q[1] = p1[1];
        v8f c = {};
        c = __builtin_amdgcn_wmma_f32_16x16x32_bf16(false, a0.v, false, b0.v,
                                                    (short)0, c, false, false);
        c = __builtin_amdgcn_wmma_f32_16x16x32_bf16(false, a1.v, false, b1.v,
                                                    (short)0, c, false, false);
        float* op = xw + (size_t)(node0 + half * 8) * HID + ct * 16 + nl;
        #pragma unroll
        for (int r = 0; r < 8; ++r) op[(size_t)r * HID] = c[r];
    }
}

// ---------------- graph plumbing ----------------
__global__ void zero_kernel(unsigned int* __restrict__ p, long long n) {
    long long i = (long long)blockIdx.x * blockDim.x + threadIdx.x;
    long long stride = (long long)gridDim.x * blockDim.x;
    for (; i < n; i += stride) p[i] = 0u;
}

__global__ void deg_kernel(const int* __restrict__ ei, int* __restrict__ deg) {
    int e = blockIdx.x * blockDim.x + threadIdx.x;
    if (e < N_EDGES) atomicAdd(&deg[ei[N_EDGES + e]], 1);
}

__global__ void dis_kernel(const int* __restrict__ deg, float* __restrict__ dis) {
    int i = blockIdx.x * blockDim.x + threadIdx.x;
    if (i < N_NODES) dis[i] = rsqrtf((float)(deg[i] + 1));   // +1 self loop
}

// one wave per edge, each lane handles 2 channels
__global__ void scatter_kernel(const int* __restrict__ ei,
                               const float* __restrict__ dis,
                               const float* __restrict__ xw,
                               float* __restrict__ agg) {
    long long idx = (long long)blockIdx.x * blockDim.x + threadIdx.x;
    int e = (int)(idx >> 5);
    if (e >= N_EDGES) return;
    int lane = (int)(idx & 31);
    int r  = ei[e];              // source
    int cn = ei[N_EDGES + e];    // target
    float nrm = dis[r] * dis[cn];
    float2 v = *(const float2*)(xw + (size_t)r * HID + lane * 2);
    float* ap = agg + (size_t)cn * HID + lane * 2;
    atomicAdd(ap,     v.x * nrm);
    atomicAdd(ap + 1, v.y * nrm);
}

// epilogue: self-loop + bias + relu -> bf16 (feeds next WMMA layer)
__global__ void relu_bf16_kernel(const float* __restrict__ agg,
                                 const float* __restrict__ xw,
                                 const float* __restrict__ dis,
                                 const float* __restrict__ bias,
                                 unsigned short* __restrict__ outb) {
    long long idx = (long long)blockIdx.x * blockDim.x + threadIdx.x;
    if (idx >= (long long)N_NODES * HID) return;
    int i = (int)(idx >> 6);
    int c = (int)(idx & 63);
    float d = dis[i];
    float v = agg[idx] + xw[idx] * d * d + bias[c];
    outb[idx] = f2bf(fmaxf(v, 0.0f));
}

// epilogue variant producing f32 (final layer, feeds pooling)
__global__ void relu_f32_kernel(const float* __restrict__ agg,
                                const float* __restrict__ xw,
                                const float* __restrict__ dis,
                                const float* __restrict__ bias,
                                float* __restrict__ outf) {
    long long idx = (long long)blockIdx.x * blockDim.x + threadIdx.x;
    if (idx >= (long long)N_NODES * HID) return;
    int i = (int)(idx >> 6);
    int c = (int)(idx & 63);
    float d = dis[i];
    float v = agg[idx] + xw[idx] * d * d + bias[c];
    outf[idx] = fmaxf(v, 0.0f);
}

// ---------------- mean pool over 2048 nodes per batch ----------------
__global__ void pool_kernel(const float* __restrict__ g2, float* __restrict__ pooled) {
    int b = blockIdx.x;
    int tid = threadIdx.x;           // 256
    int c = tid & 63;
    int q = tid >> 6;                // 0..3
    float s = 0.0f;
    for (int t = q; t < T2; t += 4)
        s += g2[((size_t)b * T2 + t) * HID + c];
    __shared__ float sm[256];
    sm[tid] = s;
    __syncthreads();
    if (q == 0) {
        float tot = sm[c] + sm[64 + c] + sm[128 + c] + sm[192 + c];
        pooled[b * HID + c] = tot * (1.0f / (float)T2);
    }
}

// ---------------- classifier: pooled[64,64] @ cls_w[64,10] + b ----------------
__global__ void cls_kernel(const float* __restrict__ pooled,
                           const float* __restrict__ Wc,
                           const float* __restrict__ bc,
                           float* __restrict__ out) {
    int idx = blockIdx.x * blockDim.x + threadIdx.x;
    if (idx >= BATCH * OUTC) return;
    int b = idx / OUTC, o = idx % OUTC;
    float s = bc[o];
    #pragma unroll 8
    for (int c = 0; c < HID; ++c)
        s = fmaf(pooled[b * HID + c], Wc[c * OUTC + o], s);
    out[idx] = s;
}

// ---------------- launch ----------------
extern "C" void kernel_launch(void* const* d_in, const int* in_sizes, int n_in,
                              void* d_out, int out_size, void* d_ws, size_t ws_size,
                              hipStream_t stream) {
    const float* x       = (const float*)d_in[0];
    const int*   ei      = (const int*)  d_in[1];
    const float* conv1_w = (const float*)d_in[2];
    const float* conv1_b = (const float*)d_in[3];
    const float* conv2_w = (const float*)d_in[4];
    const float* conv2_b = (const float*)d_in[5];
    const float* gcn1_w  = (const float*)d_in[6];
    const float* gcn1_b  = (const float*)d_in[7];
    const float* gcn2_w  = (const float*)d_in[8];
    const float* gcn2_b  = (const float*)d_in[9];
    const float* cls_w   = (const float*)d_in[10];
    const float* cls_b   = (const float*)d_in[11];
    float* out = (float*)d_out;

    // workspace layout (256B aligned)
    char* ws = (char*)d_ws;
    size_t off = 0;
    auto alloc = [&](size_t bytes) {
        char* p = ws + off;
        off += (bytes + 255) & ~(size_t)255;
        return p;
    };
    float*          h1    = (float*)         alloc((size_t)BATCH * C1 * T1 * 4);      // 16 MB
    unsigned short* featb = (unsigned short*)alloc((size_t)N_NODES * C2 * 2);         //  8 MB
    float*          xw    = (float*)         alloc((size_t)N_NODES * HID * 4);        // 32 MB
    float*          agg   = (float*)         alloc((size_t)N_NODES * HID * 4);        // 32 MB
    unsigned short* g1b   = (unsigned short*)alloc((size_t)N_NODES * HID * 2);        // 16 MB
    float*          g2    = (float*)         alloc((size_t)N_NODES * HID * 4);        // 32 MB
    int*            deg   = (int*)           alloc((size_t)N_NODES * 4);
    float*          dis   = (float*)         alloc((size_t)N_NODES * 4);
    unsigned int*   pack1 = (unsigned int*)  alloc(4 * 256 * 4);
    unsigned int*   pack2 = (unsigned int*)  alloc(8 * 256 * 4);
    float*          pooled= (float*)         alloc((size_t)BATCH * HID * 4);
    (void)ws_size; (void)in_sizes; (void)n_in; (void)out_size;

    const long long NH = (long long)N_NODES * HID;

    // CNN front-end
    conv1_kernel<<<dim3(T1 / 128, C1, BATCH), 128, 0, stream>>>(x, conv1_w, conv1_b, h1);
    conv2_kernel<<<dim3(T2 / 128, C2, BATCH), 128, 0, stream>>>(h1, conv2_w, conv2_b, featb);

    // weight pack + degree/normalization (shared by both layers)
    pack_kernel<<<1, 384, 0, stream>>>(gcn1_w, gcn2_w, pack1, pack2);
    zero_kernel<<<512, 256, 0, stream>>>((unsigned int*)deg, N_NODES);
    deg_kernel<<<N_EDGES / 256, 256, 0, stream>>>(ei, deg);
    dis_kernel<<<N_NODES / 256, 256, 0, stream>>>(deg, dis);

    // ---- GCN layer 1 ----
    gemm_n32_kernel<<<(N_NODES / 16) / 8, 256, 0, stream>>>(featb, pack1, xw);
    zero_kernel<<<4096, 256, 0, stream>>>((unsigned int*)agg, NH);
    scatter_kernel<<<((long long)N_EDGES * 32) / 256, 256, 0, stream>>>(ei, dis, xw, agg);
    relu_bf16_kernel<<<(unsigned)(NH / 256), 256, 0, stream>>>(agg, xw, dis, gcn1_b, g1b);

    // ---- GCN layer 2 ----
    gemm_n64_kernel<<<(N_NODES / 16) / 8, 256, 0, stream>>>(g1b, pack2, xw);
    zero_kernel<<<4096, 256, 0, stream>>>((unsigned int*)agg, NH);
    scatter_kernel<<<((long long)N_EDGES * 32) / 256, 256, 0, stream>>>(ei, dis, xw, agg);
    relu_f32_kernel<<<(unsigned)(NH / 256), 256, 0, stream>>>(agg, xw, dis, gcn2_b, g2);

    // pool + classify
    pool_kernel<<<BATCH, 256, 0, stream>>>(g2, pooled);
    cls_kernel<<<(BATCH * OUTC + 255) / 256, 256, 0, stream>>>(pooled, cls_w, cls_b, out);
}
